// SinkhornKnopp_2611340116618
// MI455X (gfx1250) — matrix operand order repaired and checked by reading, not compiled, not run on hardware
//
#include <hip/hip_runtime.h>
#include <hip/hip_fp16.h>

typedef __attribute__((ext_vector_type(16))) _Float16 v16h;
typedef __attribute__((ext_vector_type(8)))  _Float16 v8h;
typedef __attribute__((ext_vector_type(4)))  _Float16 v4h;
typedef __attribute__((ext_vector_type(8)))  float    v8f;
typedef __attribute__((ext_vector_type(4)))  float    v4f;

#define SK_N 8192
#define SK_NN ((size_t)SK_N * (size_t)SK_N)
#define SK_ITERS 20

// ---------------------------------------------------------------------------
// prep: Mh[i] = fp16(exp(H[i]))   (one 256MB read, 128MB write; Mh fits in L2)
// ---------------------------------------------------------------------------
__global__ void __launch_bounds__(256) sk_prep(const float* __restrict__ H,
                                               _Float16* __restrict__ Mh) {
  size_t i = ((size_t)blockIdx.x * 256 + threadIdx.x) * 4;
  v4f h = *(const v4f*)(H + i);
  v4h m;
  m[0] = (_Float16)__expf(h[0]);
  m[1] = (_Float16)__expf(h[1]);
  m[2] = (_Float16)__expf(h[2]);
  m[3] = (_Float16)__expf(h[3]);
  *(v4h*)(Mh + i) = m;
}

// init: c = 1, v = 0
__global__ void __launch_bounds__(256) sk_init_vec(float* __restrict__ c,
                                                   float* __restrict__ v) {
  int j = blockIdx.x * 256 + threadIdx.x;
  c[j] = 1.0f;
  v[j] = 0.0f;
}

// ---------------------------------------------------------------------------
// Row pass (fp16, WMMA): r = 1 / (M c)
// One workgroup = 16 rows; 8 waves split K (1024 each); D column 0 carries the
// 16 row dot-products (lanes 0 and 16 of each wave); LDS reduce + reciprocal.
// B is built branch-free: lanes 0/16 stream the c segment from LDS; all other
// lanes repeatedly read a 32-byte zero pad (stride 0), so no exec divergence.
// ---------------------------------------------------------------------------
__global__ void __launch_bounds__(256) sk_row_h16(const _Float16* __restrict__ Mh,
                                                  const float* __restrict__ c,
                                                  float* __restrict__ r) {
  __shared__ __align__(32) _Float16 lc[SK_N + 16];  // fp16 c + 32B zero pad
  __shared__ float red[8][16];

  const int tid = threadIdx.x;
  for (int j = tid; j < SK_N; j += 256) lc[j] = (_Float16)c[j];
  if (tid < 16) lc[SK_N + tid] = (_Float16)0.0f;
  __syncthreads();

  const int wave    = tid >> 5;
  const int lane    = tid & 31;
  const int rowbase = blockIdx.x * 16;
  // A k order used here (contiguous; B matches): lane<16 -> row=lane,    K: 0-15
  //                              lane>=16 -> row=lane-16, K: 16-31
  const int row  = rowbase + (lane & 15);
  const int koff = (lane >> 4) * 16;
  const _Float16* base = Mh + (size_t)row * SK_N + koff;

  const int  k0    = wave * 1024;
  const bool part  = (lane & 15) == 0;            // lanes 0 and 16 carry B
  int        boff  = part ? (k0 + (lane >> 4) * 16) : SK_N;  // else: zero pad
  const int  bstep = part ? 32 : 0;

  v8f acc = {};
  for (int k = k0; k < k0 + 1024; k += 32) {
    v8h lo = *(const v8h*)(base + k);        // 16B
    v8h hi = *(const v8h*)(base + k + 8);    // 16B
    v16h a;
#pragma unroll
    for (int h = 0; h < 8; ++h) { a[h] = lo[h]; a[h + 8] = hi[h]; }

    v16h b = *(const v16h*)(&lc[boff]);      // 32B LDS read, branch-free
    boff += bstep;

    acc = __builtin_amdgcn_wmma_f32_16x16x32_f16(false, a, false, b,
                                                 (short)0, acc, false, false);
  }

  // D column 0: lane 0 -> rows M=0..7 (VGPR 0..7), lane 16 -> rows M=8..15
  if (lane == 0) {
    for (int m = 0; m < 8; ++m) red[wave][m] = acc[m];
  } else if (lane == 16) {
    for (int m = 0; m < 8; ++m) red[wave][8 + m] = acc[m];
  }
  __syncthreads();
  if (tid < 16) {
    float s = 0.0f;
#pragma unroll
    for (int w = 0; w < 8; ++w) s += red[w][tid];
    r[rowbase + tid] = 1.0f / s;
  }
}

// ---------------------------------------------------------------------------
// Col pass (fp16): v += Mᵀ r  (coalesced column reduction; 4 cols/thread)
// grid (8 panels of 1024 cols, 64 chunks of 128 rows)
// ---------------------------------------------------------------------------
__global__ void __launch_bounds__(256) sk_col_h16(const _Float16* __restrict__ Mh,
                                                  const float* __restrict__ r,
                                                  float* __restrict__ v) {
  __shared__ float rs[128];
  const int tid  = threadIdx.x;
  const int col  = blockIdx.x * 1024 + tid * 4;
  const int row0 = blockIdx.y * 128;
  if (tid < 128) rs[tid] = r[row0 + tid];
  __syncthreads();

  float a0 = 0.f, a1 = 0.f, a2 = 0.f, a3 = 0.f;
  const _Float16* p = Mh + (size_t)row0 * SK_N + col;
#pragma unroll 4
  for (int i = 0; i < 128; ++i) {
    v4h m = *(const v4h*)p;
    p += SK_N;
    float rv = rs[i];
    a0 += (float)m[0] * rv;
    a1 += (float)m[1] * rv;
    a2 += (float)m[2] * rv;
    a3 += (float)m[3] * rv;
  }
  atomicAdd(&v[col + 0], a0);
  atomicAdd(&v[col + 1], a1);
  atomicAdd(&v[col + 2], a2);
  atomicAdd(&v[col + 3], a3);
}

// c = 1/v; re-zero v for the next accumulation
__global__ void __launch_bounds__(256) sk_recip(float* __restrict__ v,
                                                float* __restrict__ c) {
  int j = blockIdx.x * 256 + threadIdx.x;
  c[j] = 1.0f / v[j];
  v[j] = 0.0f;
}

// ---------------------------------------------------------------------------
// fp32 refinement passes (exp recomputed from H) — used for the last iteration
// (and as full fallback if workspace is too small for the fp16 matrix copy)
// ---------------------------------------------------------------------------
__global__ void __launch_bounds__(256) sk_row_f32(const float* __restrict__ H,
                                                  const float* __restrict__ c,
                                                  float* __restrict__ r) {
  const int wave = threadIdx.x >> 5;
  const int lane = threadIdx.x & 31;
  const int row  = blockIdx.x * 8 + wave;
  const float* h = H + (size_t)row * SK_N;
  float acc = 0.0f;
  for (int j = lane; j < SK_N; j += 32)
    acc += __expf(h[j]) * c[j];
#pragma unroll
  for (int off = 16; off > 0; off >>= 1)
    acc += __shfl_xor(acc, off, 32);
  if (lane == 0) r[row] = 1.0f / acc;
}

__global__ void __launch_bounds__(256) sk_col_f32(const float* __restrict__ H,
                                                  const float* __restrict__ r,
                                                  float* __restrict__ v) {
  __shared__ float rs[128];
  const int tid  = threadIdx.x;
  const int col  = blockIdx.x * 1024 + tid * 4;
  const int row0 = blockIdx.y * 128;
  if (tid < 128) rs[tid] = r[row0 + tid];
  __syncthreads();

  float a0 = 0.f, a1 = 0.f, a2 = 0.f, a3 = 0.f;
  const float* p = H + (size_t)row0 * SK_N + col;
  for (int i = 0; i < 128; ++i) {
    v4f hv = *(const v4f*)p;
    p += SK_N;
    float rv = rs[i];
    a0 += __expf(hv[0]) * rv;
    a1 += __expf(hv[1]) * rv;
    a2 += __expf(hv[2]) * rv;
    a3 += __expf(hv[3]) * rv;
  }
  atomicAdd(&v[col + 0], a0);
  atomicAdd(&v[col + 1], a1);
  atomicAdd(&v[col + 2], a2);
  atomicAdd(&v[col + 3], a3);
}

// out[i,j] = r[i] * exp(H[i,j]) * c[j]
__global__ void __launch_bounds__(256) sk_final(const float* __restrict__ H,
                                                const float* __restrict__ r,
                                                const float* __restrict__ c,
                                                float* __restrict__ out) {
  size_t i = ((size_t)blockIdx.x * 256 + threadIdx.x) * 4;
  const int row = (int)(i >> 13);          // / 8192
  const int col = (int)(i & (SK_N - 1));
  const float rv = r[row];
  v4f h = *(const v4f*)(H + i);
  v4f o;
  o[0] = rv * __expf(h[0]) * c[col + 0];
  o[1] = rv * __expf(h[1]) * c[col + 1];
  o[2] = rv * __expf(h[2]) * c[col + 2];
  o[3] = rv * __expf(h[3]) * c[col + 3];
  *(v4f*)(out + i) = o;
}

extern "C" void kernel_launch(void* const* d_in, const int* in_sizes, int n_in,
                              void* d_out, int out_size, void* d_ws, size_t ws_size,
                              hipStream_t stream) {
  (void)in_sizes; (void)n_in; (void)out_size;
  const float* H = (const float*)d_in[0];
  float* out = (float*)d_out;

  const size_t mh_bytes  = SK_NN * sizeof(_Float16);         // 128 MB
  const size_t vec_bytes = 3 * (size_t)SK_N * sizeof(float); // 96 KB
  const bool have_mh = ws_size >= mh_bytes + vec_bytes;

  char* ws = (char*)d_ws;
  _Float16* Mh = (_Float16*)ws;
  float* r;
  if (have_mh) r = (float*)(ws + mh_bytes);
  else         r = (float*)ws;
  float* c = r + SK_N;
  float* v = c + SK_N;

  const int elem_blocks = (int)(SK_NN / 4 / 256);  // 65536
  const dim3 colgrid(8, 64);

  sk_init_vec<<<SK_N / 256, 256, 0, stream>>>(c, v);

  if (have_mh) {
    sk_prep<<<elem_blocks, 256, 0, stream>>>(H, Mh);
    // iterations 1..19 on the L2-resident fp16 matrix
    for (int t = 0; t < SK_ITERS - 1; ++t) {
      sk_row_h16<<<SK_N / 16, 256, 0, stream>>>(Mh, c, r);
      sk_col_h16<<<colgrid, 256, 0, stream>>>(Mh, r, v);
      sk_recip<<<SK_N / 256, 256, 0, stream>>>(v, c);
    }
    // iteration 20 in fp32 straight from H: exact final normalization
    sk_row_f32<<<SK_N / 8, 256, 0, stream>>>(H, c, r);
    sk_col_f32<<<colgrid, 256, 0, stream>>>(H, r, v);
    sk_recip<<<SK_N / 256, 256, 0, stream>>>(v, c);
  } else {
    // workspace too small for the fp16 copy: all-fp32 path from H
    for (int t = 0; t < SK_ITERS; ++t) {
      sk_row_f32<<<SK_N / 8, 256, 0, stream>>>(H, c, r);
      sk_col_f32<<<colgrid, 256, 0, stream>>>(H, r, v);
      sk_recip<<<SK_N / 256, 256, 0, stream>>>(v, c);
    }
  }

  sk_final<<<elem_blocks, 256, 0, stream>>>(H, r, c, out);
}
